// MultiHeadAttention_25821343383700
// MI455X (gfx1250) — matrix operand order repaired and checked by reading, not compiled, not run on hardware
//
#include <hip/hip_runtime.h>

// MI455X (gfx1250) CDNA5 multi-head attention, wave32 + v_wmma_f32_16x16x32_f16.
// B=4, S=1024, E=2048, H=16, D=128.

typedef __attribute__((ext_vector_type(16))) _Float16 v16h;
typedef __attribute__((ext_vector_type(4)))  _Float16 v4h;
typedef __attribute__((ext_vector_type(8)))  float    v8f;
typedef __attribute__((ext_vector_type(4)))  float    v4f;

#define BATCH 4
#define SEQ   1024
#define EMB   2048
#define HEADS 16
#define HDIM  128

__device__ __forceinline__ v8f wmma16(v16h a, v16h b, v8f c) {
  // D = A(16x32 f16) * B(32x16 f16) + C(16x16 f32)
  return __builtin_amdgcn_wmma_f32_16x16x32_f16(
      /*neg_a=*/false, a, /*neg_b=*/false, b,
      /*c_mod=*/(short)0, c, /*reuse_a=*/false, /*reuse_b=*/false);
}

// ---- fragment loaders (per-lane base pointers, k0 advances) ----------------
// A (16x32, MxK): lane L holds row (L&15); lanes 0-15: K=[0..15];
// lanes 16-31: K=[16..31].  base = row*ld + ((lane>>4)<<4).
__device__ __forceinline__ v16h a_frag_at(const _Float16* base, int k0) {
  const _Float16* p = base + k0;
  v16h a;
#pragma unroll
  for (int i = 0; i < 8; ++i) a[i] = p[i];
#pragma unroll
  for (int i = 0; i < 8; ++i) a[8 + i] = p[8 + i];
  return a;
}

// B (32x16, KxN) from row-major [N,K]: lane L = column (L&15); lanes 0-15
// hold K=[0..15], lanes 16-31 K=[16..31].  base = col*ld + ((lane>>4)<<4).
__device__ __forceinline__ v16h b_frag_at(const _Float16* base, int k0) {
  const _Float16* p = base + k0;
  v16h b;
#pragma unroll
  for (int i = 0; i < 16; ++i) b[i] = p[i];
  return b;
}

// A-frag reload of the probability tile staged in LDS ([16 rows][32 k] f16).
__device__ __forceinline__ v16h load_a_frag_lds(const _Float16* pl, int lane) {
  const int row = lane & 15;
  const int kb  = (lane >> 4) << 4;
  const _Float16* p = pl + row * 32 + kb;
  v16h a;
#pragma unroll
  for (int i = 0; i < 8; ++i) a[i] = p[i];
#pragma unroll
  for (int i = 0; i < 8; ++i) a[8 + i] = p[8 + i];
  return a;
}

// ---- f32 -> f16 conversion pass (4 elements/thread) ------------------------
__global__ void cvt_f16(const float* __restrict__ in, _Float16* __restrict__ out) {
  const int i = blockIdx.x * blockDim.x + threadIdx.x;
  v4f v = ((const v4f*)in)[i];
  v4h h;
  h[0] = (_Float16)v[0]; h[1] = (_Float16)v[1];
  h[2] = (_Float16)v[2]; h[3] = (_Float16)v[3];
  ((v4h*)out)[i] = h;
}

// ---- GEMM: C[M,N] = A[M,K](f16) * W[N,K]^T(f16), f32 accumulate ------------
// Wave tile 64x32 (8 WMMAs / K-step); 4 waves -> block tile 128x64.
template <typename OutT>
__global__ void __launch_bounds__(128) gemm_h(const _Float16* __restrict__ A,
                                              const _Float16* __restrict__ W,
                                              OutT* __restrict__ C,
                                              int M, int N, int K) {
  const int lane = threadIdx.x & 31;
  const int wave = threadIdx.x >> 5;
  const int m0 = blockIdx.y * 128 + (wave >> 1) * 64;
  const int n0 = blockIdx.x * 64 + (wave & 1) * 32;

  const _Float16* abase[4];
#pragma unroll
  for (int mi = 0; mi < 4; ++mi)
    abase[mi] = A + (size_t)(m0 + mi * 16 + (lane & 15)) * K + ((lane >> 4) << 4);
  const _Float16* bbase[2];
#pragma unroll
  for (int ni = 0; ni < 2; ++ni)
    bbase[ni] = W + (size_t)(n0 + ni * 16 + (lane & 15)) * K + ((lane >> 4) << 4);

  v8f c[4][2];
#pragma unroll
  for (int mi = 0; mi < 4; ++mi) { c[mi][0] = v8f{}; c[mi][1] = v8f{}; }

  for (int k0 = 0; k0 < K; k0 += 32) {
    v16h b0 = b_frag_at(bbase[0], k0);
    v16h b1 = b_frag_at(bbase[1], k0);
#pragma unroll
    for (int mi = 0; mi < 4; ++mi) {
      v16h a = a_frag_at(abase[mi], k0);
      c[mi][0] = wmma16(a, b0, c[mi][0]);
      c[mi][1] = wmma16(a, b1, c[mi][1]);
    }
    if (k0 + 128 < K) {                 // gfx1250 global_prefetch_b8
      __builtin_prefetch(bbase[0] + k0 + 128, 0, 0);
      __builtin_prefetch(bbase[1] + k0 + 128, 0, 0);
#pragma unroll
      for (int mi = 0; mi < 4; ++mi)
        __builtin_prefetch(abase[mi] + k0 + 128, 0, 0);
    }
  }

  const int hi = lane >> 4, nc = lane & 15;
#pragma unroll
  for (int mi = 0; mi < 4; ++mi) {
#pragma unroll
    for (int i = 0; i < 8; ++i) {
      const int r = m0 + mi * 16 + i + 8 * hi;   // C/D layout: VGPR i -> M=i(+8)
      C[(size_t)r * N + n0 + nc]      = (OutT)c[mi][0][i];
      C[(size_t)r * N + n0 + 16 + nc] = (OutT)c[mi][1][i];
    }
  }
}

// ---- RoPE + pack to f16 [B,H,S,D] ------------------------------------------
// P: f16 [B*S, E] projection output.  scale folds 1/sqrt(HDIM) into Q.
__global__ void rope_pack(const _Float16* __restrict__ P,
                          const float* __restrict__ cosT,
                          const float* __restrict__ sinT,
                          _Float16* __restrict__ Out, float scale) {
  const int idx = blockIdx.x * blockDim.x + threadIdx.x; // B*S*H*64 threads
  const int dd = idx & 63;
  const int h  = (idx >> 6) & (HEADS - 1);
  const int s  = (idx >> 10) & (SEQ - 1);
  const int b  = idx >> 20;
  const _Float16* row = P + ((size_t)(b * SEQ + s)) * EMB + h * HDIM;
  const float x1 = (float)row[dd];
  const float x2 = (float)row[dd + 64];
  const float cv = cosT[s * HDIM + dd];  // emb = concat(freqs,freqs)
  const float sv = sinT[s * HDIM + dd];
  const float o1 = (x1 * cv - x2 * sv) * scale;
  const float o2 = (x2 * cv + x1 * sv) * scale;
  _Float16* orow = Out + ((size_t)((b * HEADS + h) * SEQ + s)) * HDIM;
  orow[dd]      = (_Float16)o1;
  orow[dd + 64] = (_Float16)o2;
}

// ---- V pack: f16 [B*S,E] -> f16 [B,H,D,S] (transposed for P·V B-frags) -----
__global__ void v_pack(const _Float16* __restrict__ P, _Float16* __restrict__ Vt) {
  const int idx = blockIdx.x * blockDim.x + threadIdx.x; // B*H*D*S threads
  const int s  = idx & (SEQ - 1);
  const int d  = (idx >> 10) & (HDIM - 1);
  const int bh = idx >> 17;
  const int b = bh >> 4, h = bh & 15;
  Vt[idx] = P[((size_t)(b * SEQ + s)) * EMB + h * HDIM + d];
}

// ---- Flash attention: one wave per 16-query tile ---------------------------
__global__ void __launch_bounds__(128) attn_kernel(const _Float16* __restrict__ Qh,
                                                   const _Float16* __restrict__ Kh,
                                                   const _Float16* __restrict__ Vt,
                                                   _Float16* __restrict__ Ctx) {
  __shared__ _Float16 pbuf[4][16 * 32];
  const int lane = threadIdx.x & 31;
  const int wave = threadIdx.x >> 5;
  const int gw = blockIdx.x * 4 + wave;          // 4096 wave-tiles total
  const int qt = gw & 63;
  const int bh = gw >> 6;
  const int q0 = qt * 16;
  const _Float16* Qp = Qh + (size_t)bh * (SEQ * HDIM);
  const _Float16* Kp = Kh + (size_t)bh * (SEQ * HDIM);
  const _Float16* Vp = Vt + (size_t)bh * (HDIM * SEQ);
  _Float16* pl = &pbuf[wave][0];

  const int hi = lane >> 4, n = lane & 15;

  // Q A-fragments across D=128 (Q pre-scaled by 1/sqrt(128))
  v16h qa[4];
#pragma unroll
  for (int t = 0; t < 4; ++t)
    qa[t] = a_frag_at(Qp + (size_t)(q0 + n) * HDIM + (hi << 4), t * 32);

  v8f o[8];
#pragma unroll
  for (int t = 0; t < 8; ++t) o[t] = v8f{};
  float rm[8], rl[8];
#pragma unroll
  for (int i = 0; i < 8; ++i) { rm[i] = -1e30f; rl[i] = 0.0f; }

  for (int k0 = 0; k0 < q0 + 16; k0 += 32) {     // causal: keys <= q0+15
    // scores 16x32 = two 16x16 C-frags
    v8f s0 = {}, s1 = {};
    const _Float16* kb0base = Kp + (size_t)(k0 + n) * HDIM + (hi << 4);
    const _Float16* kb1base = kb0base + 16 * HDIM;
#pragma unroll
    for (int t = 0; t < 4; ++t) {
      v16h kb0 = b_frag_at(kb0base, t * 32);
      v16h kb1 = b_frag_at(kb1base, t * 32);
      s0 = wmma16(qa[t], kb0, s0);
      s1 = wmma16(qa[t], kb1, s1);
    }

    // online softmax over the 16x32 tile (rows i+8*hi, cols = lane&15)
    float corr[8];
#pragma unroll
    for (int i = 0; i < 8; ++i) {
      const int q = q0 + i + 8 * hi;
      float a = (k0 + n      <= q) ? s0[i] : -1e30f;
      float b = (k0 + 16 + n <= q) ? s1[i] : -1e30f;
      float mx = fmaxf(a, b);
#pragma unroll
      for (int off = 1; off < 16; off <<= 1)
        mx = fmaxf(mx, __shfl_xor(mx, off, 32));  // stays within 16-lane half
      const float mnew = fmaxf(rm[i], mx);
      const float p0 = __expf(a - mnew);
      const float p1 = __expf(b - mnew);
      float rs = p0 + p1;
#pragma unroll
      for (int off = 1; off < 16; off <<= 1)
        rs += __shfl_xor(rs, off, 32);
      const float cf = __expf(rm[i] - mnew);
      rl[i] = rl[i] * cf + rs;
      rm[i] = mnew;
      corr[i] = cf;
      const int row = i + 8 * hi;
      pl[row * 32 + n]      = (_Float16)p0;       // stage P tile in LDS
      pl[row * 32 + 16 + n] = (_Float16)p1;
    }
#pragma unroll
    for (int t = 0; t < 8; ++t)
#pragma unroll
      for (int i = 0; i < 8; ++i) o[t][i] *= corr[i];

    // reload P in A-frag layout (per-wave in-order LDS)
    v16h pa = load_a_frag_lds(pl, lane);

    // context += P(16x32) * V(32x128): 8 WMMAs over D tiles
#pragma unroll
    for (int t = 0; t < 8; ++t) {
      const _Float16* p = Vp + (size_t)(t * 16 + n) * SEQ + k0 + (hi << 4);
      v16h vb;
#pragma unroll
      for (int i = 0; i < 16; ++i) vb[i] = p[i];
      o[t] = wmma16(pa, vb, o[t]);
    }
  }

  // normalize + write f16 context [B,S,E]
  const int b = bh >> 4, h = bh & 15;
#pragma unroll
  for (int i = 0; i < 8; ++i) {
    const int q = q0 + i + 8 * hi;
    const float inv = 1.0f / rl[i];
    _Float16* crow = Ctx + ((size_t)(b * SEQ + q)) * EMB + h * HDIM;
#pragma unroll
    for (int t = 0; t < 8; ++t) crow[t * 16 + n] = (_Float16)(o[t][i] * inv);
  }
}

extern "C" void kernel_launch(void* const* d_in, const int* in_sizes, int n_in,
                              void* d_out, int out_size, void* d_ws, size_t ws_size,
                              hipStream_t stream) {
  const float* query = (const float*)d_in[0];
  const float* keyin = (const float*)d_in[1];
  const float* value = (const float*)d_in[2];
  // d_in[3] = causal mask (unused; handled analytically)
  const float* rcos = (const float*)d_in[4];
  const float* rsin = (const float*)d_in[5];
  const float* Wq = (const float*)d_in[6];
  const float* Wk = (const float*)d_in[7];
  const float* Wv = (const float*)d_in[8];
  const float* Wo = (const float*)d_in[9];
  float* out = (float*)d_out;

  // Workspace (f16 everywhere; ~88 MB total):
  char* ws = (char*)d_ws;
  _Float16* Xh    = (_Float16*)ws;                     // 16.8 MB  input copy
  _Float16* Wh    = (_Float16*)(ws + 16777216);        //  8.4 MB  weight copy
  _Float16* projh = (_Float16*)(ws + 25165824);        // 16.8 MB  proj out / ctx
  _Float16* Qh    = (_Float16*)(ws + 41943040);        // 16.8 MB  [B,H,S,D]
  _Float16* Kh    = (_Float16*)(ws + 58720256);        // 16.8 MB  [B,H,S,D]
  _Float16* Vt    = (_Float16*)(ws + 75497472);        // 16.8 MB  [B,H,D,S]

  const int M = BATCH * SEQ;   // 4096
  const int N = EMB;           // 2048
  const int K = EMB;           // 2048
  dim3 ggrid(N / 64, M / 128); // (32, 32)
  dim3 gblk(128);
  const int in_blk = (M * K / 4) / 256;   // 8192 blocks (8.4M elems)
  const int w_blk  = (N * K / 4) / 256;   // 4096 blocks (4.2M elems)
  const int rope_blk = (BATCH * SEQ * HEADS * 64) / 256;   // 16384
  const int vt_blk   = (BATCH * HEADS * HDIM * SEQ) / 256; // 32768

  // Q projection (+RoPE, 1/sqrt(D) folded into Q)
  cvt_f16<<<in_blk, 256, 0, stream>>>(query, Xh);
  cvt_f16<<<w_blk, 256, 0, stream>>>(Wq, Wh);
  gemm_h<_Float16><<<ggrid, gblk, 0, stream>>>(Xh, Wh, projh, M, N, K);
  rope_pack<<<rope_blk, 256, 0, stream>>>(projh, rcos, rsin, Qh,
                                          0.08838834764831845f);
  // K projection (+RoPE)
  cvt_f16<<<in_blk, 256, 0, stream>>>(keyin, Xh);
  cvt_f16<<<w_blk, 256, 0, stream>>>(Wk, Wh);
  gemm_h<_Float16><<<ggrid, gblk, 0, stream>>>(Xh, Wh, projh, M, N, K);
  rope_pack<<<rope_blk, 256, 0, stream>>>(projh, rcos, rsin, Kh, 1.0f);
  // V projection (+transpose pack)
  cvt_f16<<<in_blk, 256, 0, stream>>>(value, Xh);
  cvt_f16<<<w_blk, 256, 0, stream>>>(Wv, Wh);
  gemm_h<_Float16><<<ggrid, gblk, 0, stream>>>(Xh, Wh, projh, M, N, K);
  v_pack<<<vt_blk, 256, 0, stream>>>(projh, Vt);

  // flash attention -> f16 context (reuses projh)
  attn_kernel<<<(BATCH * HEADS * (SEQ / 16)) / 4, 128, 0, stream>>>(Qh, Kh, Vt, projh);

  // output projection -> f32 d_out
  cvt_f16<<<w_blk, 256, 0, stream>>>(Wo, Wh);
  gemm_h<float><<<ggrid, gblk, 0, stream>>>(projh, Wh, out, M, N, K);
}
